// CEBlock_21672404975792
// MI455X (gfx1250) — matrix operand order repaired and checked by reading, not compiled, not run on hardware
//
#include <hip/hip_runtime.h>
#include <math.h>
#include <stdint.h>

// ---------------- problem constants ----------------
#define DIMV    768
#define HEADS   12
#define HD      64
#define NTOK    320      // 64 template + 256 search
#define TLEN    64
#define SLEN    256
#define BSZ     64
#define KEEP    180      // ceil(0.7*256)
#define NNEW    244      // 64 + 180
#define MLPH    3072
#define MTOT    (BSZ*NTOK)   // 20480
#define MNEW    (BSZ*NNEW)   // 15616
#define KCH     256          // GEMM K-chunk staged in LDS (divides 768 and 3072)

typedef __attribute__((ext_vector_type(16))) __bf16 v16bf;
typedef __attribute__((ext_vector_type(8)))  float  v8f;
typedef __attribute__((ext_vector_type(4)))  unsigned int v4u;

union Frag {
    v16bf v;
    v4u   q[2];
    unsigned short s[16];
};

__device__ __forceinline__ unsigned short f2bf(float f) {
    unsigned u = __float_as_uint(f);
    unsigned r = u + 0x7FFFu + ((u >> 16) & 1u);
    return (unsigned short)(r >> 16);
}
__device__ __forceinline__ float bf2f(unsigned short h) {
    return __uint_as_float(((unsigned)h) << 16);
}

// CDNA5 async global->LDS copy (16B per lane), tracked by ASYNCcnt.
__device__ __forceinline__ void async_b128(unsigned lds_off, const void* g) {
    asm volatile("global_load_async_to_lds_b128 %0, %1, off"
                 :: "v"(lds_off), "v"((unsigned long long)(uintptr_t)g)
                 : "memory");
}
__device__ __forceinline__ void async_wait0() {
    asm volatile("s_wait_asynccnt 0x0" ::: "memory");
}

// ---------------- utility kernels ----------------
__global__ void k_zero(float* p, int n) {
    int i = blockIdx.x * 256 + threadIdx.x;
    if (i < n) p[i] = 0.0f;
}

__global__ void k_cvt_bf16(const float* __restrict__ s, unsigned short* __restrict__ d, int n) {
    int i = blockIdx.x * 256 + threadIdx.x;
    if (i < n) d[i] = f2bf(s[i]);
}

__global__ void k_copy_i64(const long long* __restrict__ s, long long* __restrict__ d, int n) {
    int i = blockIdx.x * 256 + threadIdx.x;
    if (i < n) d[i] = s[i];
}

// ---------------- layernorm -> bf16 ----------------
__global__ void __launch_bounds__(256)
k_layernorm(const float* __restrict__ x, const float* __restrict__ g,
            const float* __restrict__ be, unsigned short* __restrict__ o) {
    int t = blockIdx.x;
    int tid = threadIdx.x;
    const float* xr = x + (size_t)t * DIMV;
    float lv[3];
    float s = 0.f, sq = 0.f;
#pragma unroll
    for (int i = 0; i < 3; i++) {
        int d = tid + i * 256;
        float v = xr[d];
        lv[i] = v; s += v; sq += v * v;
    }
    __shared__ float rs[256], rq[256];
    rs[tid] = s; rq[tid] = sq;
    __syncthreads();
    for (int st = 128; st > 0; st >>= 1) {
        if (tid < st) { rs[tid] += rs[tid + st]; rq[tid] += rq[tid + st]; }
        __syncthreads();
    }
    __shared__ float mS, rS;
    if (tid == 0) {
        float m = rs[0] / (float)DIMV;
        float var = rq[0] / (float)DIMV - m * m;
        mS = m; rS = rsqrtf(var + 1e-5f);
    }
    __syncthreads();
#pragma unroll
    for (int i = 0; i < 3; i++) {
        int d = tid + i * 256;
        o[(size_t)t * DIMV + d] = f2bf((lv[i] - mS) * rS * g[d] + be[d]);
    }
}

// ---------------- WMMA GEMM: C[M,Nout] = A[M,K] * W[Nout,K]^T ----------------
// 64x128 block tile; 8 waves, each wave: 4 m-tiles x 1 n-tile (B reused 4x).
// A staged to LDS with async double buffering (GLOBAL_LOAD_ASYNC_TO_LDS_B128).
// mode 0: f32 (+bias if non-null) ; mode 1: bf16 ; mode 2: bias + exact GELU -> bf16
__device__ __forceinline__ void epilogue_tile(v8f acc, int row0, int colg, int Nout,
                                              const float* bias, void* Cout, int mode) {
    if (mode == 0) {
        float bv = bias ? bias[colg] : 0.f;
        float* C = (float*)Cout;
#pragma unroll
        for (int r = 0; r < 8; r++)
            C[(size_t)(row0 + r) * Nout + colg] = acc[r] + bv;
    } else if (mode == 1) {
        unsigned short* C = (unsigned short*)Cout;
#pragma unroll
        for (int r = 0; r < 8; r++)
            C[(size_t)(row0 + r) * Nout + colg] = f2bf(acc[r]);
    } else {
        float bv = bias[colg];
        unsigned short* C = (unsigned short*)Cout;
#pragma unroll
        for (int r = 0; r < 8; r++) {
            float v = acc[r] + bv;
            v = 0.5f * v * (1.0f + erff(v * 0.70710678118654752f));
            C[(size_t)(row0 + r) * Nout + colg] = f2bf(v);
        }
    }
}

__global__ void __launch_bounds__(256)
k_gemm(const unsigned short* __restrict__ A, const unsigned short* __restrict__ W,
       const float* __restrict__ bias, void* __restrict__ Cout,
       int M, int K, int Nout, int mode) {
    extern __shared__ unsigned short sA[];   // 2 buffers of 64 x KCH bf16 (2 x 32KB)
    const int tid = threadIdx.x;
    const int m0 = blockIdx.y * 64;
    const unsigned ldsbase = (unsigned)(uintptr_t)(void*)sA;
    const int nch = K / KCH;

    // stage chunk 0 into buffer 0 (async path, ASYNCcnt)
    {
        const unsigned short* src = A + (size_t)m0 * K;
        for (int i = tid; i < (64 * KCH) / 8; i += 256) {
            int r = i >> 5, c = i & 31;          // 32 x 16B chunks per row
            unsigned lo = ldsbase + (unsigned)(r * KCH + c * 8) * 2u;
            async_b128(lo, src + (size_t)r * K + c * 8);
        }
        async_wait0();
    }
    __syncthreads();

    const int wave = tid >> 5, lane = tid & 31;
    const int hf = lane >> 4, ln = lane & 15;
    const int n0 = blockIdx.x * 128 + wave * 16;
    const unsigned short* wrow = W + (size_t)(n0 + ln) * K;

    v8f acc0 = {}, acc1 = {}, acc2 = {}, acc3 = {};
    for (int c = 0; c < nch; c++) {
        const unsigned short* cur = sA + (c & 1) * (64 * KCH);
        // overlap: issue async stage of chunk c+1 into the other buffer,
        // and prefetch the next W chunk (global_prefetch_b8)
        if (c + 1 < nch) {
            const unsigned short* src = A + (size_t)m0 * K + (c + 1) * KCH;
            unsigned nbase = ldsbase + (unsigned)(((c + 1) & 1) * (64 * KCH)) * 2u;
            for (int i = tid; i < (64 * KCH) / 8; i += 256) {
                int r = i >> 5, cc = i & 31;
                unsigned lo = nbase + (unsigned)(r * KCH + cc * 8) * 2u;
                async_b128(lo, src + (size_t)r * K + cc * 8);
            }
            __builtin_prefetch(wrow + (c + 1) * KCH, 0, 0);
            __builtin_prefetch(wrow + (c + 1) * KCH + 128, 0, 0);
        }
        const unsigned short* wch = wrow + c * KCH;
#pragma unroll 4
        for (int kb = 0; kb < KCH; kb += 32) {
            // load ALL fragments first so the ds/global loads clause together,
            // then fire 4 independent WMMAs back-to-back
            Frag b, a0, a1, a2, a3;
            const unsigned short* bp = wch + kb + hf * 8;
            b.q[0] = *(const v4u*)bp;
            b.q[1] = *(const v4u*)(bp + 16);
            const unsigned short* ap0 = cur + (0 * 16 + ln) * KCH + kb + hf * 8;
            const unsigned short* ap1 = cur + (1 * 16 + ln) * KCH + kb + hf * 8;
            const unsigned short* ap2 = cur + (2 * 16 + ln) * KCH + kb + hf * 8;
            const unsigned short* ap3 = cur + (3 * 16 + ln) * KCH + kb + hf * 8;
            a0.q[0] = *(const v4u*)ap0; a0.q[1] = *(const v4u*)(ap0 + 16);
            a1.q[0] = *(const v4u*)ap1; a1.q[1] = *(const v4u*)(ap1 + 16);
            a2.q[0] = *(const v4u*)ap2; a2.q[1] = *(const v4u*)(ap2 + 16);
            a3.q[0] = *(const v4u*)ap3; a3.q[1] = *(const v4u*)(ap3 + 16);
            acc0 = __builtin_amdgcn_wmma_f32_16x16x32_bf16(false, a0.v, false, b.v,
                                                           (short)0, acc0, false, false);
            acc1 = __builtin_amdgcn_wmma_f32_16x16x32_bf16(false, a1.v, false, b.v,
                                                           (short)0, acc1, false, false);
            acc2 = __builtin_amdgcn_wmma_f32_16x16x32_bf16(false, a2.v, false, b.v,
                                                           (short)0, acc2, false, false);
            acc3 = __builtin_amdgcn_wmma_f32_16x16x32_bf16(false, a3.v, false, b.v,
                                                           (short)0, acc3, false, false);
        }
        async_wait0();
        __syncthreads();
    }

    const int colg = n0 + ln;
    epilogue_tile(acc0, m0 + 0  + hf * 8, colg, Nout, bias, Cout, mode);
    epilogue_tile(acc1, m0 + 16 + hf * 8, colg, Nout, bias, Cout, mode);
    epilogue_tile(acc2, m0 + 32 + hf * 8, colg, Nout, bias, Cout, mode);
    epilogue_tile(acc3, m0 + 48 + hf * 8, colg, Nout, bias, Cout, mode);
}

// ---------------- attention: per (qtile, head, batch) ----------------
// LDS: sQ 64x64 bf16 | sS 64x320 f32 | sP 64x320 bf16 | sVt 64x320 bf16 (V^T)
__global__ void __launch_bounds__(256)
k_attn(const unsigned short* __restrict__ qkv, float* __restrict__ attn_out,
       float* __restrict__ attn_t, unsigned short* __restrict__ ctx) {
    extern __shared__ char smem[];
    unsigned short* sQ  = (unsigned short*)smem;                                   // 8KB
    float*          sS  = (float*)(smem + 64 * 64 * 2);                            // 80KB
    unsigned short* sP  = (unsigned short*)(smem + 64 * 64 * 2 + 64 * NTOK * 4);   // 40KB
    unsigned short* sVt = (unsigned short*)(smem + 64 * 64 * 2 + 64 * NTOK * 4
                                                 + 64 * NTOK * 2);                 // 40KB

    const int qt = blockIdx.x, h = blockIdx.y, b = blockIdx.z;
    const int tid = threadIdx.x, wave = tid >> 5, lane = tid & 31;
    const int hf = lane >> 4, ln = lane & 15;
    const int q0 = qt * 64;
    const size_t rstr = 3 * DIMV;  // 2304
    const unsigned short* qbase = qkv + (size_t)b * NTOK * rstr + h * HD;

    // stage Q tile (64 rows x 64 bf16) via async-to-LDS
    {
        unsigned qlds = (unsigned)(uintptr_t)(void*)sQ;
        for (int i = tid; i < 512; i += 256) {
            int r = i >> 3, c = i & 7;
            async_b128(qlds + (unsigned)i * 16u,
                       qbase + (size_t)(q0 + r) * rstr + c * 8);
        }
        async_wait0();
    }
    // stage V^T (sVt[d][key] = V[key][d]) with coalesced 128-bit reads
    {
        const unsigned short* vbase = qbase + 2 * DIMV;
        for (int i = tid; i < NTOK * 8; i += 256) {      // 320 rows x 8 chunks
            int key = i >> 3, c = i & 7;
            v4u raw = *(const v4u*)(vbase + (size_t)key * rstr + c * 8);
            union { v4u q; unsigned short s[8]; } u; u.q = raw;
#pragma unroll
            for (int j = 0; j < 8; j++)
                sVt[(c * 8 + j) * NTOK + key] = u.s[j];
        }
    }
    __syncthreads();

    // scores: 4x20 tiles of 16x16, 10 per wave; K streamed from global
#pragma unroll 1
    for (int t = 0; t < 10; t++) {
        int tIdx = wave + (t << 3);
        int mt = tIdx / 20, nt = tIdx % 20;
        v8f acc = {};
#pragma unroll
        for (int kb = 0; kb < HD; kb += 32) {
            Frag a, bf;
            const unsigned short* ap = sQ + (mt * 16 + ln) * HD + kb + hf * 8;
            a.q[0] = *(const v4u*)ap;
            a.q[1] = *(const v4u*)(ap + 16);
            const unsigned short* bp = qbase + DIMV + (size_t)(nt * 16 + ln) * rstr + kb + hf * 8;
            bf.q[0] = *(const v4u*)bp;
            bf.q[1] = *(const v4u*)(bp + 16);
            acc = __builtin_amdgcn_wmma_f32_16x16x32_bf16(false, a.v, false, bf.v,
                                                          (short)0, acc, false, false);
        }
#pragma unroll
        for (int r = 0; r < 8; r++)
            sS[(mt * 16 + hf * 8 + r) * NTOK + nt * 16 + ln] = acc[r] * 0.125f;
    }
    __syncthreads();

    // softmax: 8 rows per wave, wave32 shuffle reductions
    float* arow_base = attn_out + (((size_t)b * HEADS + h) * NTOK + q0) * NTOK;
#pragma unroll 1
    for (int rr = 0; rr < 8; rr++) {
        int row = wave * 8 + rr;
        float vals[10];
        float mx = -3.0e38f;
#pragma unroll
        for (int i = 0; i < 10; i++) {
            vals[i] = sS[row * NTOK + lane + i * 32];
            mx = fmaxf(mx, vals[i]);
        }
        for (int o = 16; o > 0; o >>= 1) mx = fmaxf(mx, __shfl_xor(mx, o));
        float sum = 0.f;
#pragma unroll
        for (int i = 0; i < 10; i++) { vals[i] = __expf(vals[i] - mx); sum += vals[i]; }
        for (int o = 16; o > 0; o >>= 1) sum += __shfl_xor(sum, o);
        float inv = 1.0f / sum;
#pragma unroll
        for (int i = 0; i < 10; i++) {
            int col = lane + i * 32;
            float p = vals[i] * inv;
            arow_base[(size_t)row * NTOK + col] = p;
            sP[row * NTOK + col] = f2bf(p);
        }
    }
    __syncthreads();

    // pruning score accumulation: template queries live only in qtile 0
    if (qt == 0) {
        int j = tid;  // search column 0..255
        float s = 0.f;
        for (int r = 0; r < TLEN; r++) s += bf2f(sP[r * NTOK + TLEN + j]);
        atomicAdd(&attn_t[b * SLEN + j], s);
    }

    // ctx = P(64x320) x V(320x64): 16 tiles, 2 per wave; B frags from staged V^T
#pragma unroll 1
    for (int t = 0; t < 2; t++) {
        int tIdx = wave + t * 8;
        int mt = tIdx >> 2, nt = tIdx & 3;
        v8f acc = {};
#pragma unroll 2
        for (int kc = 0; kc < 10; kc++) {
            int kb = kc * 32;
            Frag a, bf;
            const unsigned short* ap = sP + (mt * 16 + ln) * NTOK + kb + hf * 8;
            a.q[0] = *(const v4u*)ap;
            a.q[1] = *(const v4u*)(ap + 16);
            const unsigned short* bp = sVt + (nt * 16 + ln) * NTOK + kb + hf * 8;
            bf.q[0] = *(const v4u*)bp;
            bf.q[1] = *(const v4u*)(bp + 16);
            acc = __builtin_amdgcn_wmma_f32_16x16x32_bf16(false, a.v, false, bf.v,
                                                          (short)0, acc, false, false);
        }
#pragma unroll
        for (int r = 0; r < 8; r++) {
            int row = q0 + mt * 16 + hf * 8 + r;
            ctx[((size_t)b * NTOK + row) * DIMV + h * HD + nt * 16 + ln] = f2bf(acc[r]);
        }
    }
}

// ---------------- FNA adapter + residual: out = xb + fna(y) ----------------
struct F4 { float x, y, z, w; };
__global__ void __launch_bounds__(256)
k_fna_residual(const float* __restrict__ xb, const float* __restrict__ y,
               float* __restrict__ out,
               const float* __restrict__ route_w, const float* __restrict__ alpha,
               const float* __restrict__ dyt_w, const float* __restrict__ dyt_b,
               const float* __restrict__ Wa, const float* __restrict__ ba,
               const float* __restrict__ Wm, const float* __restrict__ bm,
               const float* __restrict__ Wb, const float* __restrict__ bb) {
    int t = blockIdx.x;
    int tid = threadIdx.x;
    const float* yr = y + (size_t)t * DIMV;
    float al = alpha[0];
    float p0 = 0.f, p1 = 0.f, p2 = 0.f, p3 = 0.f;
#pragma unroll
    for (int i = 0; i < 3; i++) {
        int d = tid + i * 256;
        float gt = tanhf(al * yr[d] * route_w[d]) * dyt_w[d] + dyt_b[d];
        p0 += gt * Wa[0 * DIMV + d];
        p1 += gt * Wa[1 * DIMV + d];
        p2 += gt * Wa[2 * DIMV + d];
        p3 += gt * Wa[3 * DIMV + d];
    }
    __shared__ F4 red[256];
    red[tid].x = p0; red[tid].y = p1; red[tid].z = p2; red[tid].w = p3;
    __syncthreads();
    for (int s = 128; s > 0; s >>= 1) {
        if (tid < s) {
            red[tid].x += red[tid + s].x; red[tid].y += red[tid + s].y;
            red[tid].z += red[tid + s].z; red[tid].w += red[tid + s].w;
        }
        __syncthreads();
    }
    __shared__ float xc[4];
    if (tid == 0) {
        float xa[4];
        xa[0] = fmaxf(red[0].x + ba[0], 0.f);
        xa[1] = fmaxf(red[0].y + ba[1], 0.f);
        xa[2] = fmaxf(red[0].z + ba[2], 0.f);
        xa[3] = fmaxf(red[0].w + ba[3], 0.f);
        for (int g = 0; g < 2; g++)
            for (int o = 0; o < 2; o++) {
                int idx = g * 2 + o;
                float xm = xa[g * 2 + 0] * Wm[idx * 2 + 0] + xa[g * 2 + 1] * Wm[idx * 2 + 1] + bm[idx];
                xc[idx] = fmaxf(xa[idx] + xm, 0.f);
            }
    }
    __syncthreads();
#pragma unroll
    for (int i = 0; i < 3; i++) {
        int d = tid + i * 256;
        int g = d / 384;
        float xup = xc[g * 2 + 0] * Wb[d * 2 + 0] + xc[g * 2 + 1] * Wb[d * 2 + 1] + bb[d];
        out[(size_t)t * DIMV + d] = xb[(size_t)t * DIMV + d] + xup + yr[d];
    }
}

// ---------------- top-K selection (stable rank, matches argsort(-a)) ----------------
__global__ void __launch_bounds__(256)
k_select(const float* __restrict__ attn_t, const long long* __restrict__ gi_s,
         int* __restrict__ topk, long long* __restrict__ keep_out,
         long long* __restrict__ removed_out) {
    int b = blockIdx.x, tid = threadIdx.x;
    __shared__ float a[256];
    a[tid] = attn_t[b * SLEN + tid];
    __syncthreads();
    float mine = a[tid];
    int rank = 0;
    for (int i = 0; i < SLEN; i++) {
        float v = a[i];
        rank += (v > mine) || (v == mine && i < tid);
    }
    long long g = gi_s[(size_t)b * SLEN + tid];
    if (rank < KEEP) {
        topk[b * KEEP + rank] = tid;
        keep_out[(size_t)b * KEEP + rank] = g;
    } else {
        removed_out[(size_t)b * (SLEN - KEEP) + (rank - KEEP)] = g;
    }
}

// ---------------- gather kept tokens ----------------
__global__ void k_gather(const float* __restrict__ x_res, const int* __restrict__ topk,
                         float* __restrict__ x_ce) {
    size_t i = (size_t)blockIdx.x * 256 + threadIdx.x;
    const size_t total = (size_t)MNEW * DIMV;
    if (i >= total) return;
    int d = (int)(i % DIMV);
    size_t t = i / DIMV;
    int n = (int)(t % NNEW);
    int b = (int)(t / NNEW);
    int src = (n < TLEN) ? n : TLEN + topk[b * KEEP + (n - TLEN)];
    x_ce[i] = x_res[((size_t)b * NTOK + src) * DIMV + d];
}

// ---------------- host launcher ----------------
extern "C" void kernel_launch(void* const* d_in, const int* in_sizes, int n_in,
                              void* d_out, int out_size, void* d_ws, size_t ws_size,
                              hipStream_t stream) {
    (void)n_in; (void)out_size; (void)ws_size; (void)in_sizes;
    const float*     x      = (const float*)d_in[0];
    const long long* gi_t   = (const long long*)d_in[1];
    const long long* gi_s   = (const long long*)d_in[2];
    const float*     ln1_g  = (const float*)d_in[3];
    const float*     ln1_b  = (const float*)d_in[4];
    const float*     Wqkv   = (const float*)d_in[5];
    const float*     Wproj  = (const float*)d_in[6];
    const float*     bproj  = (const float*)d_in[7];
    const float*     ln2_g  = (const float*)d_in[8];
    const float*     ln2_b  = (const float*)d_in[9];
    const float*     W1     = (const float*)d_in[10];
    const float*     b1     = (const float*)d_in[11];
    const float*     W2     = (const float*)d_in[12];
    const float*     b2     = (const float*)d_in[13];
    const float* ar[10]; for (int i = 0; i < 10; i++) ar[i] = (const float*)d_in[14 + i];
    const float* mr[10]; for (int i = 0; i < 10; i++) mr[i] = (const float*)d_in[24 + i];

    char* ws = (char*)d_ws;
    size_t off = 0;
    auto take = [&](size_t bytes) -> char* {
        char* p = ws + off;
        off += (bytes + 255) & ~(size_t)255;
        return p;
    };
    unsigned short* wqkv_bf = (unsigned short*)take((size_t)3 * DIMV * DIMV * 2);
    unsigned short* wproj_bf= (unsigned short*)take((size_t)DIMV * DIMV * 2);
    unsigned short* w1_bf   = (unsigned short*)take((size_t)MLPH * DIMV * 2);
    unsigned short* w2_bf   = (unsigned short*)take((size_t)DIMV * MLPH * 2);
    unsigned short* xn1     = (unsigned short*)take((size_t)MTOT * DIMV * 2);
    unsigned short* qkv     = (unsigned short*)take((size_t)MTOT * 3 * DIMV * 2);
    unsigned short* ctx     = (unsigned short*)take((size_t)MTOT * DIMV * 2);
    float*          f32a    = (float*)take((size_t)MTOT * DIMV * 4);   // xattn, then h2
    float*          x_res   = (float*)take((size_t)MTOT * DIMV * 4);
    float*          attn_t  = (float*)take((size_t)BSZ * SLEN * 4);
    int*            topk    = (int*)take((size_t)BSZ * KEEP * 4);
    float*          x_ce    = (float*)take((size_t)MNEW * DIMV * 4);
    unsigned short* xn2     = (unsigned short*)take((size_t)MNEW * DIMV * 2);
    unsigned short* hbuf    = (unsigned short*)take((size_t)MNEW * MLPH * 2);

    char* outp = (char*)d_out;
    const size_t X_BYTES   = (size_t)MNEW * DIMV * 4;
    const size_t GIT_BYTES = (size_t)BSZ * TLEN * 8;
    const size_t KEEP_BYTES= (size_t)BSZ * KEEP * 8;
    const size_t REM_BYTES = (size_t)BSZ * (SLEN - KEEP) * 8;
    float*     out_x    = (float*)outp;
    long long* out_git  = (long long*)(outp + X_BYTES);
    long long* out_keep = (long long*)(outp + X_BYTES + GIT_BYTES);
    long long* out_rem  = (long long*)(outp + X_BYTES + GIT_BYTES + KEEP_BYTES);
    float*     out_attn = (float*)(outp + X_BYTES + GIT_BYTES + KEEP_BYTES + REM_BYTES);

    const size_t GEMM_SMEM = (size_t)2 * 64 * KCH * 2;   // 64KB double buffer
    const size_t ATTN_SMEM = (size_t)64 * 64 * 2 + 64 * NTOK * 4
                           + 64 * NTOK * 2 + 64 * NTOK * 2;   // 168KB

    // 0) zero pruning accumulator
    {
        int n = BSZ * SLEN;
        k_zero<<<(n + 255) / 256, 256, 0, stream>>>(attn_t, n);
    }
    // 1) weights -> bf16
    {
        int n;
        n = 3 * DIMV * DIMV; k_cvt_bf16<<<(n + 255) / 256, 256, 0, stream>>>(Wqkv, wqkv_bf, n);
        n = DIMV * DIMV;     k_cvt_bf16<<<(n + 255) / 256, 256, 0, stream>>>(Wproj, wproj_bf, n);
        n = MLPH * DIMV;     k_cvt_bf16<<<(n + 255) / 256, 256, 0, stream>>>(W1, w1_bf, n);
        n = DIMV * MLPH;     k_cvt_bf16<<<(n + 255) / 256, 256, 0, stream>>>(W2, w2_bf, n);
    }
    // 2) LN1
    k_layernorm<<<MTOT, 256, 0, stream>>>(x, ln1_g, ln1_b, xn1);
    // 3) QKV GEMM (bf16 out)
    k_gemm<<<dim3((3 * DIMV) / 128, MTOT / 64), 256, GEMM_SMEM, stream>>>(
        xn1, wqkv_bf, nullptr, qkv, MTOT, DIMV, 3 * DIMV, 1);
    // 4) attention (probs -> out_attn, ctx bf16, pruning accumulation)
    k_attn<<<dim3(NTOK / 64, HEADS, BSZ), 256, ATTN_SMEM, stream>>>(
        qkv, out_attn, attn_t, ctx);
    // 5) proj GEMM (f32 + bias) -> xattn in f32a
    k_gemm<<<dim3(DIMV / 128, MTOT / 64), 256, GEMM_SMEM, stream>>>(
        ctx, wproj_bf, bproj, f32a, MTOT, DIMV, DIMV, 0);
    // 6) x_res = x + fna(xattn)
    k_fna_residual<<<MTOT, 256, 0, stream>>>(x, f32a, x_res,
        ar[0], ar[1], ar[2], ar[3], ar[4], ar[5], ar[6], ar[7], ar[8], ar[9]);
    // 7) top-K selection + index outputs
    k_select<<<BSZ, 256, 0, stream>>>(attn_t, gi_s, topk, out_keep, out_rem);
    // 8) gather kept tokens
    {
        size_t total = (size_t)MNEW * DIMV;
        k_gather<<<(unsigned)((total + 255) / 256), 256, 0, stream>>>(x_res, topk, x_ce);
    }
    // 9) LN2
    k_layernorm<<<MNEW, 256, 0, stream>>>(x_ce, ln2_g, ln2_b, xn2);
    // 10) MLP1: GELU(xn2 @ W1^T + b1) -> bf16
    k_gemm<<<dim3(MLPH / 128, MNEW / 64), 256, GEMM_SMEM, stream>>>(
        xn2, w1_bf, b1, hbuf, MNEW, DIMV, MLPH, 2);
    // 11) MLP2: h @ W2^T + b2 -> f32 (reuse f32a)
    k_gemm<<<dim3(DIMV / 128, MNEW / 64), 256, GEMM_SMEM, stream>>>(
        hbuf, w2_bf, b2, f32a, MNEW, MLPH, DIMV, 0);
    // 12) final: out_x = x_ce + fna(h2)
    k_fna_residual<<<MNEW, 256, 0, stream>>>(x_ce, f32a, out_x,
        mr[0], mr[1], mr[2], mr[3], mr[4], mr[5], mr[6], mr[7], mr[8], mr[9]);
    // 13) pass-through template indices
    k_copy_i64<<<(BSZ * TLEN + 255) / 256, 256, 0, stream>>>(gi_t, out_git, BSZ * TLEN);
}
